// SAST_Net_7954279432540
// MI455X (gfx1250) — matrix operand order repaired and checked by reading, not compiled, not run on hardware
//
#include <hip/hip_runtime.h>
#include <hip/hip_bf16.h>
#include <hip/hip_fp16.h>

typedef __attribute__((ext_vector_type(16))) _Float16 v16h;
typedef __attribute__((ext_vector_type(8)))  _Float16 v8h;
typedef __attribute__((ext_vector_type(8)))  float    v8f;
typedef __attribute__((ext_vector_type(4)))  float    v4f;

#define BSZ 8
#define NN  512
#define DD  64
#define NKEEP 256
#define RSTR 36   // padded row stride (words) for conflict-free LDS transpose

// Branch-free tanh: prefer CDNA5 hardware V_TANH_F32, else 1 - 2/(exp(2x)+1)
static __device__ __forceinline__ float fast_tanh(float x) {
#if __has_builtin(__builtin_amdgcn_tanhf)
  return __builtin_amdgcn_tanhf(x);
#else
  float e = __expf(2.0f * x);
  return 1.0f - 2.0f / (e + 1.0f);
#endif
}

// ---------------------------------------------------------------------------
// Kernel 0: convert x, att_w, pwa_w, pwo_w to f16 staging buffers
// ---------------------------------------------------------------------------
__global__ __launch_bounds__(256) void k_convert(
    const float* __restrict__ x, const float* __restrict__ aw,
    const float* __restrict__ wa, const float* __restrict__ wo,
    _Float16* __restrict__ xh, _Float16* __restrict__ awh,
    _Float16* __restrict__ wah, _Float16* __restrict__ woh)
{
  int e = blockIdx.x * 256 + threadIdx.x;
  if (e < BSZ * NN * DD) xh[e] = (_Float16)x[e];
  if (e < DD * DD) {
    awh[e] = (_Float16)aw[e];
    wah[e] = (_Float16)wa[e];
    woh[e] = (_Float16)wo[e];
  }
}

// ---------------------------------------------------------------------------
// Kernel 1: per (b,i) wave: logits over all j via WMMA, hw-tanh, att_weight
//           dot (LDS transpose-reduce), softmax over j, attention row in f16.
//   logits[b,i,j] = att_weight . tanh(att_w^T (x_i*x_j) + att_b) / TEMP
// ---------------------------------------------------------------------------
__global__ __launch_bounds__(256) void k_logits_softmax(
    const _Float16* __restrict__ xh, const _Float16* __restrict__ wh,
    const float* __restrict__ att_b, const float* __restrict__ att_weight,
    _Float16* __restrict__ attnh)
{
  __shared__ __align__(16) _Float16 sW[DD * DD];     // 8 KB
  __shared__ __align__(16) float sRed[8][16 * RSTR]; // 18 KB transpose pad
  const int tid = threadIdx.x;
  const int wave = tid >> 5, lane = tid & 31;

  for (int q = 0; q < 16; ++q) sW[tid + 256 * q] = wh[tid + 256 * q];
  __syncthreads();

  const int bi = blockIdx.x * 8 + wave;  // 0..4095
  const int b = bi >> 9, i = bi & 511;
  const int m = lane & 15, hf = lane >> 4;
  const int n = m;

  // B fragments: B[k,e] = W[k, et*16+n]; lane n = column, K = hf*8 + h + (h>>3)*8
  v16h Bf[2][4];
  for (int kc = 0; kc < 2; ++kc)
    for (int et = 0; et < 4; ++et) {
      v16h t;
      for (int h = 0; h < 16; ++h)
        t[h] = sW[(kc * 32 + hf * 8 + h + (h >> 3) * 8) * DD + et * 16 + n];
      Bf[kc][et] = t;
    }

  float awv[4], bbv[4];
  for (int et = 0; et < 4; ++et) {
    awv[et] = att_weight[et * 16 + n] * 0.5f;  // fold 1/TEMP (TEMP=2)
    bbv[et] = att_b[et * 16 + n];
  }

  // x_i k-segments (A layout: per lane two contiguous 8-half runs per k-chunk)
  const _Float16* xib = xh + (size_t)(b * NN + i) * DD;
  v8h xi0[2], xi1[2];
  for (int kc = 0; kc < 2; ++kc) {
    xi0[kc] = *(const v8h*)(xib + kc * 32 + hf * 8);
    xi1[kc] = *(const v8h*)(xib + kc * 32 + 16 + hf * 8);
  }

  float l[16];
  for (int t = 0; t < 16; ++t) l[t] = 0.f;

  const float* rrow = &sRed[wave][(lane & 15) * RSTR];

  for (int jt = 0; jt < 32; ++jt) {
    const _Float16* xjb = xh + (size_t)(b * NN + jt * 16 + m) * DD;
    v16h A[2];
    for (int kc = 0; kc < 2; ++kc) {
      v8h j0 = *(const v8h*)(xjb + kc * 32 + hf * 8);
      v8h j1 = *(const v8h*)(xjb + kc * 32 + 16 + hf * 8);
      v8h a0 = j0 * xi0[kc];
      v8h a1 = j1 * xi1[kc];
      v16h a;
      for (int q = 0; q < 8; ++q) { a[q] = a0[q]; a[8 + q] = a1[q]; }
      A[kc] = a;
    }
    v8f dacc = {};
    for (int et = 0; et < 4; ++et) {
      v8f c = {};
      c = __builtin_amdgcn_wmma_f32_16x16x32_f16(false, A[0], false, Bf[0][et],
                                                 (short)0, c, false, false);
      c = __builtin_amdgcn_wmma_f32_16x16x32_f16(false, A[1], false, Bf[1][et],
                                                 (short)0, c, false, false);
      for (int r = 0; r < 8; ++r)
        dacc[r] += fast_tanh(c[r] + bbv[et]) * awv[et];
    }
    // transpose-reduce over N via LDS:
    // dacc[r] = C[M = r + 8*hf][N = lane&15] -> sRed[row=M][col=N]
    for (int r = 0; r < 8; ++r)
      sRed[wave][(hf * 8 + r) * RSTR + n] = dacc[r];
    // lane reads row j = lane&15 (16 floats, conflict-free b128 x4), sums
    v4f s0 = *(const v4f*)(rrow + 0);
    v4f s1 = *(const v4f*)(rrow + 4);
    v4f s2 = *(const v4f*)(rrow + 8);
    v4f s3 = *(const v4f*)(rrow + 12);
    float lg = ((s0[0] + s0[1]) + (s0[2] + s0[3])) +
               ((s1[0] + s1[1]) + (s1[2] + s1[3])) +
               ((s2[0] + s2[1]) + (s2[2] + s2[3])) +
               ((s3[0] + s3[1]) + (s3[2] + s3[3]));
    // pack: logit[jt*16 + (lane&15)]; register slot t=jt/2, half selects jt&1
    if ((jt & 1) == hf) l[jt >> 1] = lg;
  }

  // wave-local softmax over the 512 logits: l[t] = logit[32*t + lane]
  float mx = -1e30f;
  for (int t = 0; t < 16; ++t) mx = fmaxf(mx, l[t]);
  for (int msk = 1; msk < 32; msk <<= 1) mx = fmaxf(mx, __shfl_xor(mx, msk, 32));
  float sum = 0.f;
  for (int t = 0; t < 16; ++t) { l[t] = __expf(l[t] - mx); sum += l[t]; }
  for (int msk = 1; msk < 32; msk <<= 1) sum += __shfl_xor(sum, msk, 32);
  const float inv = 1.f / sum;
  _Float16* arow = attnh + (size_t)(b * NN + i) * NN;
  for (int t = 0; t < 16; ++t) arow[lane + 32 * t] = (_Float16)(l[t] * inv);
}

// ---------------------------------------------------------------------------
// Kernel 2: agg = attn @ X (WMMA), h = agg@pwa + x@pwo + biases (WMMA),
//           BN(eval) + SELU, score = sigmoid(h@pool_w + b), hs = h*score.
// One wave per (b, i-tile of 16).
// ---------------------------------------------------------------------------
__global__ __launch_bounds__(256) void k_agg_h(
    const _Float16* __restrict__ xh, const _Float16* __restrict__ attnh,
    const _Float16* __restrict__ wah, const _Float16* __restrict__ woh,
    const float* __restrict__ pwa_b, const float* __restrict__ pwo_b,
    const float* __restrict__ gamma, const float* __restrict__ beta,
    const float* __restrict__ rmean, const float* __restrict__ rvar,
    const float* __restrict__ pool_w, const float* __restrict__ pool_b,
    float* __restrict__ hsbuf, float* __restrict__ scorebuf)
{
  __shared__ __align__(16) _Float16 sWa[DD * DD];        // 8 KB
  __shared__ __align__(16) _Float16 sWo[DD * DD];        // 8 KB
  __shared__ __align__(16) _Float16 sAgg[8][16 * DD];    // 16 KB
  const int tid = threadIdx.x;
  const int wave = tid >> 5, lane = tid & 31;
  for (int q = 0; q < 16; ++q) {
    sWa[tid + 256 * q] = wah[tid + 256 * q];
    sWo[tid + 256 * q] = woh[tid + 256 * q];
  }
  __syncthreads();

  const int task = blockIdx.x * 8 + wave;  // 0..255
  const int b = task >> 5, it = task & 31;
  const int m = lane & 15, hf = lane >> 4;
  const int n = m;

  // Phase 1: agg tile (16 rows x 64 cols) = attn[rows, :] @ X[b]
  for (int et = 0; et < 4; ++et) {
    v8f c = {};
    for (int kc = 0; kc < 16; ++kc) {
      const _Float16* arow =
          attnh + (size_t)(b * NN + it * 16 + m) * NN + kc * 32;
      v8h a0 = *(const v8h*)(arow + hf * 8);
      v8h a1 = *(const v8h*)(arow + 16 + hf * 8);
      v16h A;
      for (int q = 0; q < 8; ++q) { A[q] = a0[q]; A[8 + q] = a1[q]; }
      v16h Bf;
      const _Float16* xb =
          xh + (size_t)(b * NN + kc * 32) * DD + et * 16 + n;
      for (int h = 0; h < 16; ++h) Bf[h] = xb[(size_t)(hf * 8 + h + (h >> 3) * 8) * DD];
      c = __builtin_amdgcn_wmma_f32_16x16x32_f16(false, A, false, Bf, (short)0,
                                                 c, false, false);
    }
    for (int r = 0; r < 8; ++r)
      sAgg[wave][(r + 8 * hf) * DD + et * 16 + n] = (_Float16)c[r];
  }

  // Phase 2: h = agg@pwa + x@pwo + biases; BN; SELU
  const _Float16* xrow = xh + (size_t)(b * NN + it * 16 + m) * DD;
  v16h XA[2], GA[2];
  for (int kc = 0; kc < 2; ++kc) {
    v8h a0 = *(const v8h*)(xrow + kc * 32 + hf * 8);
    v8h a1 = *(const v8h*)(xrow + kc * 32 + 16 + hf * 8);
    v16h A;
    for (int q = 0; q < 8; ++q) { A[q] = a0[q]; A[8 + q] = a1[q]; }
    XA[kc] = A;
    const _Float16* grow = &sAgg[wave][m * DD + kc * 32];
    v8h g0 = *(const v8h*)(grow + hf * 8);
    v8h g1 = *(const v8h*)(grow + 16 + hf * 8);
    v16h G;
    for (int q = 0; q < 8; ++q) { G[q] = g0[q]; G[8 + q] = g1[q]; }
    GA[kc] = G;
  }

  v8f hh[4];
  for (int et = 0; et < 4; ++et) {
    v8f c = {};
    for (int kc = 0; kc < 2; ++kc) {
      v16h Ba, Bo;
      for (int h = 0; h < 16; ++h) {
        int kk = kc * 32 + hf * 8 + h + (h >> 3) * 8;
        Ba[h] = sWa[kk * DD + et * 16 + n];
        Bo[h] = sWo[kk * DD + et * 16 + n];
      }
      c = __builtin_amdgcn_wmma_f32_16x16x32_f16(false, GA[kc], false, Ba,
                                                 (short)0, c, false, false);
      c = __builtin_amdgcn_wmma_f32_16x16x32_f16(false, XA[kc], false, Bo,
                                                 (short)0, c, false, false);
    }
    const int d = et * 16 + n;
    const float bias = pwa_b[d] + pwo_b[d];
    const float scl = gamma[d] * rsqrtf(rvar[d] + 1e-5f);
    const float sft = beta[d] - rmean[d] * scl;
    v8f hv;
    for (int r = 0; r < 8; ++r) {
      float v = (c[r] + bias) * scl + sft;
      v = 1.0507009873554805f *
          (v > 0.f ? v : 1.6732632423543772f * (__expf(v) - 1.f));
      hv[r] = v;
    }
    hh[et] = hv;
  }

  // Phase 3: pooling score per row + hs = h*score
  float pw[4];
  for (int et = 0; et < 4; ++et) pw[et] = pool_w[et * 16 + n];
  const float pb = pool_b[0];
  for (int r = 0; r < 8; ++r) {
    float s = 0.f;
    for (int et = 0; et < 4; ++et) s += hh[et][r] * pw[et];
    s += __shfl_xor(s, 1, 32);
    s += __shfl_xor(s, 2, 32);
    s += __shfl_xor(s, 4, 32);
    s += __shfl_xor(s, 8, 32);
    const float sc = 1.f / (1.f + __expf(-(s + pb)));
    const int row = it * 16 + r + 8 * hf;
    if (n == 0) scorebuf[b * NN + row] = sc;
    float* hrow = hsbuf + (size_t)(b * NN + row) * DD;
    for (int et = 0; et < 4; ++et) hrow[et * 16 + n] = hh[et][r] * sc;
  }
}

// ---------------------------------------------------------------------------
// Kernel 3: per-batch top-k (stable, descending) + gather of hs rows
// ---------------------------------------------------------------------------
__global__ __launch_bounds__(512) void k_topk(
    const float* __restrict__ scorebuf, const float* __restrict__ hsbuf,
    float* __restrict__ out)
{
  __shared__ float s[NN];
  __shared__ int idx[NKEEP];
  const int b = blockIdx.x, t = threadIdx.x;
  s[t] = scorebuf[b * NN + t];
  __syncthreads();
  const float mys = s[t];
  int rank = 0;
  for (int j = 0; j < NN; ++j) {
    float sj = s[j];
    rank += (sj > mys) || (sj == mys && j < t);
  }
  if (rank < NKEEP) idx[rank] = t;
  __syncthreads();
  for (int q = 0; q < 32; ++q) {
    int e = t + 512 * q;  // 0 .. 16383
    int r = e >> 6, dd = e & 63;
    out[(size_t)b * (NKEEP * DD) + e] =
        hsbuf[((size_t)b * NN + idx[r]) * DD + dd];
  }
}

// ---------------------------------------------------------------------------
extern "C" void kernel_launch(void* const* d_in, const int* in_sizes, int n_in,
                              void* d_out, int out_size, void* d_ws,
                              size_t ws_size, hipStream_t stream) {
  const float* x     = (const float*)d_in[0];
  const float* aw    = (const float*)d_in[1];   // att_w (d,d)
  const float* ab    = (const float*)d_in[2];   // att_b (d,)
  const float* awt   = (const float*)d_in[3];   // att_weight (d,1)
  const float* pwa_w = (const float*)d_in[4];
  const float* pwa_b = (const float*)d_in[5];
  const float* pwo_w = (const float*)d_in[6];
  const float* pwo_b = (const float*)d_in[7];
  const float* gamma = (const float*)d_in[8];
  const float* beta  = (const float*)d_in[9];
  const float* rmean = (const float*)d_in[10];
  const float* rvar  = (const float*)d_in[11];
  const float* pool_w = (const float*)d_in[12];
  const float* pool_b = (const float*)d_in[13];
  float* out = (float*)d_out;

  // workspace layout (bytes)
  char* ws = (char*)d_ws;
  _Float16* xh    = (_Float16*)(ws + 0);                    // 512 KB
  _Float16* awh   = (_Float16*)(ws + 524288);               //   8 KB
  _Float16* wah   = (_Float16*)(ws + 532480);               //   8 KB
  _Float16* woh   = (_Float16*)(ws + 540672);               //   8 KB
  _Float16* attnh = (_Float16*)(ws + 548864);               //   4 MB
  float*    hsbuf = (float*)   (ws + 4743168);              //   1 MB
  float*    score = (float*)   (ws + 5791744);              //  16 KB

  k_convert<<<1024, 256, 0, stream>>>(x, aw, pwa_w, pwo_w, xh, awh, wah, woh);
  k_logits_softmax<<<512, 256, 0, stream>>>(xh, awh, ab, awt, attnh);
  k_agg_h<<<32, 256, 0, stream>>>(xh, attnh, wah, woh, pwa_b, pwo_b, gamma,
                                  beta, rmean, rvar, pool_w, pool_b, hsbuf,
                                  score);
  k_topk<<<8, 512, 0, stream>>>(score, hsbuf, out);
}
